// CrossAttentionMatchup_26164940767438
// MI455X (gfx1250) — matrix operand order repaired and checked by reading, not compiled, not run on hardware
//
#include <hip/hip_runtime.h>
#include <hip/hip_bf16.h>

// Problem constants (from reference)
#define EMBED   256
#define NHEAD   8
#define HDIM    32
#define NQ      64
#define NKV     64
#define NBATCH  2048
#define LN_EPS  1e-6f
#define NEG_BIG (-1000000000.0f)

typedef __attribute__((ext_vector_type(16))) _Float16 v16h;
typedef __attribute__((ext_vector_type(8)))  _Float16 v8h;
typedef __attribute__((ext_vector_type(4)))  _Float16 v4h;
typedef __attribute__((ext_vector_type(8)))  float    v8f;
typedef __attribute__((ext_vector_type(4)))  float    v4f;

// ---------------------------------------------------------------------------
// WMMA fragment helpers (CDNA5 wave32 layouts, CDNA5 ISA §7.12.2)
// A 16x32 f16: lanes 0-15 = rows M, VGPR0-3 K=0..7, VGPR4-7 K=16..23;
//              lanes 16-31 same rows, K offset +8.
// B 32x16 f16: lane&15 = column N. Loaders here use K = 16*(lane>>4) + s for both A and B.
// C/D 16x16 f32: lane&15 = column N; VGPR r = row r (+8 for lanes 16-31).
// All loaders below are contiguous vector loads (16B/32B aligned).
// ---------------------------------------------------------------------------

// A fragment from row-major f16 (ld = row stride). Two 16B loads + concat.
__device__ __forceinline__ v16h wmma_load_A_h(const _Float16* base, int ld,
                                              int mBase, int kBase, int lane) {
  const int row  = mBase + (lane & 15);
  const int koff = (lane >> 4) << 4;                  // 0 or 16
  const _Float16* p = base + row * ld + kBase + koff;
  v8h lo = *(const v8h*)(p);                          // K = koff + 0..7
  v8h hi = *(const v8h*)(p + 8);                      // K = koff + 8..15
  return __builtin_shufflevector(lo, hi, 0, 1, 2, 3, 4, 5, 6, 7,
                                 8, 9, 10, 11, 12, 13, 14, 15);
}

// A fragment from row-major f32 with on-the-fly f16 conversion (4x 16B loads).
__device__ __forceinline__ v16h wmma_load_A_f32(const float* base, int ld,
                                                int mBase, int kBase, int lane) {
  const int row  = mBase + (lane & 15);
  const int koff = (lane >> 4) << 4;
  const float* p = base + row * ld + kBase + koff;
  v4f x0 = *(const v4f*)(p);
  v4f x1 = *(const v4f*)(p + 4);
  v4f x2 = *(const v4f*)(p + 8);
  v4f x3 = *(const v4f*)(p + 12);
  v16h a;
#pragma unroll
  for (int j = 0; j < 4; ++j) {
    a[j]      = (_Float16)x0[j];
    a[4 + j]  = (_Float16)x1[j];
    a[8 + j]  = (_Float16)x2[j];
    a[12 + j] = (_Float16)x3[j];
  }
  return a;
}

// B fragment from an N-major (transposed) weight matrix: baseT[n*256 + k].
// Lane n reads 16 contiguous halves (32B, 32B-aligned) -> 2x global_load_b128.
__device__ __forceinline__ v16h wmma_load_B_wT(const _Float16* baseT,
                                               int kBase, int nBase, int lane) {
  const int n  = nBase + (lane & 15);
  const int k0 = kBase + ((lane >> 4) << 4);          // 0 or 16
  return *(const v16h*)(baseT + n * EMBED + k0);
}

// B fragment for q@k^T: B[k][n] = sK[n][colBase + k]; contiguous per lane.
__device__ __forceinline__ v16h wmma_load_B_kT(const _Float16* sK,
                                               int colBase, int nBase, int lane) {
  const int n  = nBase + (lane & 15);
  const int k0 = (lane >> 4) << 4;
  return *(const v16h*)(sK + n * EMBED + colBase + k0);
}

// B fragment for attn@V from dim-major V: sVt[d*64 + kv]; contiguous per lane.
__device__ __forceinline__ v16h wmma_load_B_vT(const _Float16* sVt,
                                               int kBase, int dBase, int lane) {
  const int d  = dBase + (lane & 15);
  const int k0 = kBase + ((lane >> 4) << 4);
  return *(const v16h*)(sVt + d * NKV + k0);
}

__device__ __forceinline__ v8f wmma_f16(v16h a, v16h b, v8f c) {
  // (neg_a, A, neg_b, B, c_mod, C, reuse_a, reuse_b)
  return __builtin_amdgcn_wmma_f32_16x16x32_f16(false, a, false, b,
                                                (short)0, c, false, false);
}

// ---------------------------------------------------------------------------
// Kernel 0: convert weights to f16, TRANSPOSED to N-major, into workspace.
// Layout (halves): [0,65536) WqT[256n][256k] | [65536,196608) WkvT[512n][256k]
//                  | [196608,262144) WoT[256n][256k]
// ---------------------------------------------------------------------------
__global__ void cvt_weights_kernel(const float* __restrict__ Wq,
                                   const float* __restrict__ Wkv,
                                   const float* __restrict__ Wo,
                                   _Float16* __restrict__ w16) {
  const int i      = blockIdx.x * blockDim.x + threadIdx.x;
  const int stride = gridDim.x * blockDim.x;
  for (int j = i; j < EMBED * EMBED; j += stride) {     // j = n*256 + k
    const int n = j >> 8, k = j & 255;
    w16[j] = (_Float16)Wq[k * EMBED + n];
  }
  for (int j = i; j < 2 * EMBED * EMBED; j += stride) { // j = n*256 + k, n<512
    const int n = j >> 8, k = j & 255;
    w16[65536 + j] = (_Float16)Wkv[k * (2 * EMBED) + n];
  }
  for (int j = i; j < EMBED * EMBED; j += stride) {
    const int n = j >> 8, k = j & 255;
    w16[65536 + 131072 + j] = (_Float16)Wo[k * EMBED + n];
  }
}

// ---------------------------------------------------------------------------
// Main kernel: one block (8 wave32) per batch element.
// LDS (288 KB dynamic):
//   [  0K, 32K) sQ   f16 64x256   row-major   (projected Q)
//   [ 32K, 64K) sK   f16 64x256   row-major   (projected K)
//   [ 64K, 96K) sVt  f16 256x64   dim-major   (projected V, transposed)
//   [ 96K,160K) sO   f32 64x256               (attention output)
//   [160K,288K) sS   f32 8x64x64              (per-head scores)
//      overlays: [160K,192K) sAst f16 query staging
//                [192K,224K) sBst f16 key_value staging
//                [160K,224K) sX   f32 64x256 LayerNorm input
// ---------------------------------------------------------------------------
__global__ __launch_bounds__(256, 1)
void xattn_ln_kernel(const float* __restrict__ query,
                     const float* __restrict__ key_value,
                     const int*   __restrict__ kv_mask,
                     const _Float16* __restrict__ w16,
                     const float* __restrict__ bq,
                     const float* __restrict__ bkv,
                     const float* __restrict__ bo,
                     const float* __restrict__ ln_gamma,
                     const float* __restrict__ ln_beta,
                     float* __restrict__ out) {
  extern __shared__ char smem[];
  _Float16* sQ   = (_Float16*)(smem);
  _Float16* sK   = (_Float16*)(smem + 32 * 1024);
  _Float16* sVt  = (_Float16*)(smem + 64 * 1024);
  float*    sO   = (float*)   (smem + 96 * 1024);
  float*    sS   = (float*)   (smem + 160 * 1024);
  _Float16* sAst = (_Float16*)(smem + 160 * 1024);
  _Float16* sBst = (_Float16*)(smem + 192 * 1024);
  float*    sX   = (float*)   (smem + 160 * 1024);

  const int b    = blockIdx.x;
  const int tid  = threadIdx.x;
  const int wave = tid >> 5;
  const int lane = tid & 31;

  const float* qg  = query     + (size_t)b * (NQ * EMBED);
  const float* kvg = key_value + (size_t)b * (NKV * EMBED);
  const _Float16* WqT  = w16;
  const _Float16* WkvT = w16 + 65536;
  const _Float16* WoT  = w16 + 65536 + 131072;

  // Warm the weight path toward L2 (global_prefetch_b8)
#pragma unroll
  for (int i = 0; i < 4; ++i)
    __builtin_prefetch((const char*)w16 + (((tid << 2) + i) << 9), 0, 2);

  // ---- P0: stage inputs as f16 into LDS (vectorized b128 in / b64 out) -
  for (int i = tid; i < (NQ * EMBED) / 4; i += 256) {
    v4f x = ((const v4f*)qg)[i];
    v4h y;
#pragma unroll
    for (int j = 0; j < 4; ++j) y[j] = (_Float16)x[j];
    ((v4h*)sAst)[i] = y;
  }
  for (int i = tid; i < (NKV * EMBED) / 4; i += 256) {
    v4f x = ((const v4f*)kvg)[i];
    v4h y;
#pragma unroll
    for (int j = 0; j < 4; ++j) y[j] = (_Float16)x[j];
    ((v4h*)sBst)[i] = y;
  }
  __syncthreads();

  // ---- P1: q = query @ Wq + bq (64x256); wave owns 2 N-tiles ----------
  for (int nn = 0; nn < 2; ++nn) {
    const int nt = wave * 2 + nn;
    v8f acc[4] = {};
    for (int kk = 0; kk < 8; ++kk) {
      v16h bb = wmma_load_B_wT(WqT, kk * 32, nt * 16, lane);  // loaded once
#pragma unroll
      for (int mt = 0; mt < 4; ++mt) {
        v16h a = wmma_load_A_h(sAst, EMBED, mt * 16, kk * 32, lane);
        acc[mt] = wmma_f16(a, bb, acc[mt]);
      }
    }
    const int n    = nt * 16 + (lane & 15);
    const int moff = (lane >> 4) << 3;
    const float bias = bq[n];
#pragma unroll
    for (int mt = 0; mt < 4; ++mt)
#pragma unroll
      for (int r = 0; r < 8; ++r)
        sQ[(mt * 16 + moff + r) * EMBED + n] = (_Float16)(acc[mt][r] + bias);
  }

  // ---- P2: kv = key_value @ Wkv + bkv (64x512 -> sK | sVt) ------------
  for (int nn = 0; nn < 4; ++nn) {
    const int nt = wave * 4 + nn;          // 0..31 over 512 columns
    v8f acc[4] = {};
    for (int kk = 0; kk < 8; ++kk) {
      v16h bb = wmma_load_B_wT(WkvT, kk * 32, nt * 16, lane);
#pragma unroll
      for (int mt = 0; mt < 4; ++mt) {
        v16h a = wmma_load_A_h(sBst, EMBED, mt * 16, kk * 32, lane);
        acc[mt] = wmma_f16(a, bb, acc[mt]);
      }
    }
    const int n    = nt * 16 + (lane & 15);
    const int moff = (lane >> 4) << 3;
    const float bias = bkv[n];
    if (n < EMBED) {
      // K: row-major (scalar strided stores)
#pragma unroll
      for (int mt = 0; mt < 4; ++mt)
#pragma unroll
        for (int r = 0; r < 8; ++r)
          sK[(mt * 16 + moff + r) * EMBED + n] = (_Float16)(acc[mt][r] + bias);
    } else {
      // V: dim-major -> 8 contiguous halves per tile = one 16B ds_store
      const int d = n - EMBED;
#pragma unroll
      for (int mt = 0; mt < 4; ++mt) {
        v8h pk;
#pragma unroll
        for (int r = 0; r < 8; ++r) pk[r] = (_Float16)(acc[mt][r] + bias);
        *(v8h*)(sVt + d * NKV + mt * 16 + moff) = pk;
      }
    }
  }
  __syncthreads();

  // ---- P3: attention, one head per wave -------------------------------
  {
    const int h = wave;
    float* sSh = sS + h * (NQ * NKV);
    const float scale = 0.17677669529663687f; // 1/sqrt(HDIM)

    // scores = (q_h @ k_h^T) * scale + mask; B fragment reused over mt
    for (int nt = 0; nt < 4; ++nt) {
      v16h bb = wmma_load_B_kT(sK, h * HDIM, nt * 16, lane);
      const int n = nt * 16 + (lane & 15);
      const float maskadd = kv_mask[b * NKV + n] ? 0.0f : NEG_BIG;
#pragma unroll
      for (int mt = 0; mt < 4; ++mt) {
        v16h a = wmma_load_A_h(sQ, EMBED, mt * 16, h * HDIM, lane);
        v8f acc = {};
        acc = wmma_f16(a, bb, acc);
        const int m0 = mt * 16 + ((lane >> 4) << 3);
#pragma unroll
        for (int r = 0; r < 8; ++r)
          sSh[(m0 + r) * NKV + n] = acc[r] * scale + maskadd;
      }
    }

    // row softmax (LDS ops of one wave are in program order -> safe)
    for (int rr = lane; rr < NQ; rr += 32) {
      float* row = sSh + rr * NKV;
      float mx = -3.4e38f;
      for (int j = 0; j < NKV; ++j) mx = fmaxf(mx, row[j]);
      float s = 0.f;
      for (int j = 0; j < NKV; ++j) { float e = __expf(row[j] - mx); row[j] = e; s += e; }
      const float inv = 1.f / s;
      for (int j = 0; j < NKV; ++j) row[j] *= inv;
    }

    // out_h = attn @ v_h (64x32); B fragments from dim-major V, reused
    for (int nt2 = 0; nt2 < 2; ++nt2) {
      const int dBase = h * HDIM + nt2 * 16;
      v8f acc[4] = {};
      for (int kk = 0; kk < 2; ++kk) {
        v16h bb = wmma_load_B_vT(sVt, kk * 32, dBase, lane);
#pragma unroll
        for (int mt = 0; mt < 4; ++mt) {
          v16h a = wmma_load_A_f32(sSh, NKV, mt * 16, kk * 32, lane);
          acc[mt] = wmma_f16(a, bb, acc[mt]);
        }
      }
      const int n    = dBase + (lane & 15);
      const int moff = (lane >> 4) << 3;
#pragma unroll
      for (int mt = 0; mt < 4; ++mt)
#pragma unroll
        for (int r = 0; r < 8; ++r)
          sO[(mt * 16 + moff + r) * EMBED + n] = acc[mt][r];
    }
  }
  __syncthreads();

  // ---- P4: x = attn_out @ Wo + bo + query -----------------------------
  for (int nn = 0; nn < 2; ++nn) {
    const int nt = wave * 2 + nn;
    v8f acc[4] = {};
    for (int kk = 0; kk < 8; ++kk) {
      v16h bb = wmma_load_B_wT(WoT, kk * 32, nt * 16, lane);
#pragma unroll
      for (int mt = 0; mt < 4; ++mt) {
        v16h a = wmma_load_A_f32(sO, EMBED, mt * 16, kk * 32, lane);
        acc[mt] = wmma_f16(a, bb, acc[mt]);
      }
    }
    const int n    = nt * 16 + (lane & 15);
    const int moff = (lane >> 4) << 3;
    const float bias = bo[n];
#pragma unroll
    for (int mt = 0; mt < 4; ++mt)
#pragma unroll
      for (int r = 0; r < 8; ++r) {
        const int m = mt * 16 + moff + r;
        sX[m * EMBED + n] = acc[mt][r] + bias + qg[m * EMBED + n];
      }
  }
  __syncthreads();

  // ---- P5: LayerNorm, one row per thread ------------------------------
  if (tid < NQ) {
    const int row = tid;
    const float* x = sX + row * EMBED;
    float mu = 0.f;
    for (int c = 0; c < EMBED; ++c) mu += x[c];
    mu *= (1.0f / EMBED);
    float var = 0.f;
    for (int c = 0; c < EMBED; ++c) { const float d = x[c] - mu; var += d * d; }
    var *= (1.0f / EMBED);
    const float inv = rsqrtf(var + LN_EPS);
    float* o = out + (size_t)b * (NQ * EMBED) + (size_t)row * EMBED;
    for (int c = 0; c < EMBED; ++c)
      o[c] = (x[c] - mu) * inv * ln_gamma[c] + ln_beta[c];
  }
}

// ---------------------------------------------------------------------------
extern "C" void kernel_launch(void* const* d_in, const int* in_sizes, int n_in,
                              void* d_out, int out_size, void* d_ws, size_t ws_size,
                              hipStream_t stream) {
  const float* query     = (const float*)d_in[0];
  const float* key_value = (const float*)d_in[1];
  const int*   kv_mask   = (const int*)d_in[2];
  const float* Wq        = (const float*)d_in[3];
  const float* bq        = (const float*)d_in[4];
  const float* Wkv       = (const float*)d_in[5];
  const float* bkv       = (const float*)d_in[6];
  const float* Wo        = (const float*)d_in[7];
  const float* bo        = (const float*)d_in[8];
  const float* ln_gamma  = (const float*)d_in[9];
  const float* ln_beta   = (const float*)d_in[10];

  _Float16* w16 = (_Float16*)d_ws;  // 512 KB of transposed f16 weights

  cvt_weights_kernel<<<dim3(256), dim3(256), 0, stream>>>(Wq, Wkv, Wo, w16);

  const size_t smem_bytes = 288 * 1024;  // fits CDNA5's 320 KB WGP LDS
  xattn_ln_kernel<<<dim3(NBATCH), dim3(256), smem_bytes, stream>>>(
      query, key_value, kv_mask, w16, bq, bkv, bo, ln_gamma, ln_beta,
      (float*)d_out);
}
